// MultiHeadSelfAttention_24739011625678
// MI455X (gfx1250) — matrix operand order repaired and checked by reading, not compiled, not run on hardware
//
#include <hip/hip_runtime.h>
#include <hip/hip_bf16.h>

// ---------------------------------------------------------------------------
// MHA forward for MI455X (gfx1250, wave32, WMMA).
// All GEMMs in bf16 with f32 accumulation via v_wmma_f32_16x16x32_bf16.
// Weights and K are pre-transposed; B-fragment loads are k-strided, A-fragment loads contiguous
// (merged ds_load_b128); tiles are staged with global_load_async_to_lds_b128.
// ---------------------------------------------------------------------------

typedef __bf16 bf16_t;
typedef __attribute__((ext_vector_type(16))) __bf16 v16bf;
typedef __attribute__((ext_vector_type(4)))  __bf16 v4bf;
typedef __attribute__((ext_vector_type(8)))  float  v8f;

#define B_SZ   2
#define S_SZ   2048
#define D_SZ   1024
#define H_SZ   16
#define DK_SZ  64
#define M_TOT  (B_SZ * S_SZ)          // 4096

// ---- optional CDNA5 async global->LDS path --------------------------------
#if defined(__has_builtin)
#if __has_builtin(__builtin_amdgcn_global_load_async_to_lds_b128)
#define USE_ASYNC_LDS 1
#endif
#endif

#ifdef USE_ASYNC_LDS
typedef int v4i_t __attribute__((__vector_size__(16)));
typedef __attribute__((address_space(1))) v4i_t* g_v4i_p;
typedef __attribute__((address_space(3))) v4i_t* l_v4i_p;
__device__ __forceinline__ void async_copy16(const void* g, void* l) {
  __builtin_amdgcn_global_load_async_to_lds_b128((g_v4i_p)(void*)g,
                                                 (l_v4i_p)l, 0, 0);
}
__device__ __forceinline__ void async_wait0() {
  asm volatile("s_wait_asynccnt 0x0" ::: "memory");
}
#else
__device__ __forceinline__ void async_copy16(const void* g, void* l) {
  *(uint4*)l = *(const uint4*)g;
}
__device__ __forceinline__ void async_wait0() {}
#endif

// ---- fragment helpers (CDNA5 WMMA VGPR layouts, ISA 7.12.2) ---------------

// A-matrix 16x32 bf16: lane m (lane&15), khalf=lane>>4.
// VGPR j<4 holds K = 2j + 8*khalf ; VGPR j>=4 holds K = 16 + 2(j-4) + 8*khalf.
// Per lane this is two contiguous 8-half runs -> merges into 2x ds_load_b128.
__device__ __forceinline__ v16bf load_a_frag(const bf16_t* row, int khalf) {
  v16bf a;
#pragma unroll
  for (int j = 0; j < 8; ++j) {
    int k = (j < 4 ? 2 * j : 16 + 2 * (j - 4)) + khalf * 8;
    a[2 * j]     = row[k];
    a[2 * j + 1] = row[k + 1];
  }
  return a;
}

// B-matrix 32x16 bf16: lane l holds column N=l&15; K in same order as A-frag.
__device__ __forceinline__ v16bf load_b_frag(const bf16_t* p, int ld, int khalf) {
  v16bf b;
#pragma unroll
  for (int i = 0; i < 16; ++i) b[i] = p[((i < 8 ? i : 16 + (i - 8)) + khalf * 8) * ld];
  return b;
}

__device__ __forceinline__ v8f wmma_bf16(v16bf a, v16bf b, v8f c) {
  return __builtin_amdgcn_wmma_f32_16x16x32_bf16(false, a, false, b,
                                                 (short)0, c, false, false);
}

// ---------------------------------------------------------------------------
// Kernel 1a: fp32 -> bf16 convert (x), 4 elements/thread
// ---------------------------------------------------------------------------
__global__ void cvt_f32_bf16_kernel(const float* __restrict__ src,
                                    bf16_t* __restrict__ dst, int n4) {
  int i = blockIdx.x * blockDim.x + threadIdx.x;
  if (i < n4) {
    const float4 f = ((const float4*)src)[i];
    v4bf o;
    o[0] = (bf16_t)f.x; o[1] = (bf16_t)f.y;
    o[2] = (bf16_t)f.z; o[3] = (bf16_t)f.w;
    ((v4bf*)dst)[i] = o;
  }
}

// ---------------------------------------------------------------------------
// Kernel 1b: weight convert + transpose: W[n][k] f32 -> Wt[k][n] bf16.
// One-time scattered stores; GEMM B-fragments are then read k-strided.
// ---------------------------------------------------------------------------
__global__ void cvt_w_transpose_kernel(const float* __restrict__ W,
                                       bf16_t* __restrict__ Wt) {
  const int t  = blockIdx.x * blockDim.x + threadIdx.x;  // < 1024*1024/4
  const int n  = t >> 8;           // 0..1023
  const int k0 = (t & 255) * 4;    // 0..1020
  const float4 f = *(const float4*)(W + (size_t)n * D_SZ + k0);
  Wt[(size_t)(k0 + 0) * D_SZ + n] = (bf16_t)f.x;
  Wt[(size_t)(k0 + 1) * D_SZ + n] = (bf16_t)f.y;
  Wt[(size_t)(k0 + 2) * D_SZ + n] = (bf16_t)f.z;
  Wt[(size_t)(k0 + 3) * D_SZ + n] = (bf16_t)f.w;
}

// ---------------------------------------------------------------------------
// Kernel 2: C[4096 x 1024] (f32) = A[4096 x 1024] (bf16) * Wt[k][n] (bf16)
// Workgroup tile 128x128, 8 waves (2x4), wave tile 64x32, K-step 32.
// LDS double-buffered; slab t+1 staged async during compute of slab t.
// blockIdx.z selects Wt slice and C slice (fused Q/K/V projection).
// ---------------------------------------------------------------------------
__global__ __launch_bounds__(256) void gemm_bf16_kernel(
    const bf16_t* __restrict__ A, const bf16_t* __restrict__ Wtbase,
    float* __restrict__ Cbase) {
  constexpr int KK = D_SZ;   // 1024
  constexpr int NN = D_SZ;   // 1024
  constexpr int NSLAB = KK / 32;

  __shared__ __attribute__((aligned(16))) bf16_t Alds[2][128][40];
  __shared__ __attribute__((aligned(16))) bf16_t Wtl[2][32][136];

  const bf16_t* Wt = Wtbase + (size_t)blockIdx.z * KK * NN;
  float*        C  = Cbase + (size_t)blockIdx.z * (size_t)M_TOT * NN;

  const int tid   = threadIdx.x;
  const int lane  = tid & 31;
  const int wave  = tid >> 5;
  const int wm    = wave >> 2;     // 0..1
  const int wn    = wave & 3;      // 0..3
  const int m0    = blockIdx.y * 128;
  const int n0    = blockIdx.x * 128;
  const int mlane = lane & 15;
  const int khalf = lane >> 4;

  v8f acc[4][2];
  const v8f vzero = {0.f, 0.f, 0.f, 0.f, 0.f, 0.f, 0.f, 0.f};
#pragma unroll
  for (int mt = 0; mt < 4; ++mt)
#pragma unroll
    for (int nt = 0; nt < 2; ++nt) acc[mt][nt] = vzero;

  // A staging: thread copies 32B = 16 halves of one A row slab
  const int arow  = tid >> 1;      // 0..127
  const int ahalf = tid & 1;
  const bf16_t* aptr = A + (size_t)(m0 + arow) * KK + ahalf * 16;
  // Wt staging: thread copies 32B = 16 halves of one Wt row (k-row) slab
  const int wrow = tid >> 3;       // 0..31  (k within slab)
  const int wseg = tid & 7;        // 8 x 16-half segments cover 128 n
  const bf16_t* wptr = Wt + (size_t)wrow * NN + n0 + wseg * 16;

  auto stage = [&](int buf, int k0) {
    async_copy16(aptr + k0,     &Alds[buf][arow][ahalf * 16]);
    async_copy16(aptr + k0 + 8, &Alds[buf][arow][ahalf * 16 + 8]);
    async_copy16(wptr + (size_t)k0 * NN,     &Wtl[buf][wrow][wseg * 16]);
    async_copy16(wptr + (size_t)k0 * NN + 8, &Wtl[buf][wrow][wseg * 16 + 8]);
  };

  stage(0, 0);
  int cur = 0;
  for (int it = 0; it < NSLAB; ++it) {
    async_wait0();
    __syncthreads();
    if (it + 1 < NSLAB) stage(cur ^ 1, (it + 1) * 32);

    // B-frag: lane l holds B column N=l&15; K in A-frag order (strided)
    v16bf bfr[2];
#pragma unroll
    for (int nt = 0; nt < 2; ++nt)
      bfr[nt] = load_b_frag(&Wtl[cur][0][wn * 32 + nt * 16 + mlane], 136, khalf);

#pragma unroll
    for (int mt = 0; mt < 4; ++mt) {
      v16bf afr = load_a_frag(&Alds[cur][wm * 64 + mt * 16 + mlane][0], khalf);
#pragma unroll
      for (int nt = 0; nt < 2; ++nt)
        acc[mt][nt] = wmma_bf16(afr, bfr[nt], acc[mt][nt]);
    }
    cur ^= 1;
  }

  // C/D layout: lane<16 -> M=j, N=lane ; lane>=16 -> M=8+j, N=lane-16.
#pragma unroll
  for (int mt = 0; mt < 4; ++mt) {
    const int mrow0 = m0 + wm * 64 + mt * 16 + khalf * 8;
#pragma unroll
    for (int nt = 0; nt < 2; ++nt) {
      const int col = n0 + wn * 32 + nt * 16 + mlane;
#pragma unroll
      for (int j = 0; j < 8; ++j)
        C[(size_t)((mrow0 + j) * NN + col)] = acc[mt][nt][j];
    }
  }
}

// ---------------------------------------------------------------------------
// Kernel 3: RoPE on Q/K + repack to bf16:
//   Q -> [b][h][s][dk]   (A-side of QK^T)
//   K -> [b][h][dk][s]   (transposed: B-side of QK^T, fragments read d-strided)
//   V -> [b][h][s][dk]   (B-side of PV, fragments read kv-strided)
// blockIdx.y = 0(Q),1(K),2(V, no rope). One thread per channel pair.
// ---------------------------------------------------------------------------
__global__ void rope_pack_kernel(const float* __restrict__ QKVf,
                                 const int* __restrict__ pos,
                                 bf16_t* __restrict__ Qb,
                                 bf16_t* __restrict__ Kbt,
                                 bf16_t* __restrict__ Vb) {
  const int z   = blockIdx.y;
  const int idx = blockIdx.x * blockDim.x + threadIdx.x;  // < B*S*D/2 = 2^21
  const int p = idx & 31;            // pair index within head (dk/2 = 32)
  const int h = (idx >> 5) & 15;
  const int s = (idx >> 9) & 2047;
  const int b = idx >> 20;
  const int bh = b * H_SZ + h;

  const float* src = QKVf + (size_t)z * M_TOT * D_SZ +
                     (size_t)(b * S_SZ + s) * D_SZ + h * DK_SZ + 2 * p;
  const float x1 = src[0], x2 = src[1];

  if (z == 2) {
    bf16_t* dst = Vb + ((size_t)bh * S_SZ + s) * DK_SZ + 2 * p;
    dst[0] = (bf16_t)x1;
    dst[1] = (bf16_t)x2;
    return;
  }

  // inv_freq = theta^(-2p/64) = exp(-p * ln(10000)/32)
  const float inv_freq = __expf(-(float)p * 0.28782313662425572f);
  const float ang = (float)pos[b * S_SZ + s] * inv_freq;
  float sn, cs;
  sincosf(ang, &sn, &cs);
  const float r1 = x1 * cs - x2 * sn;
  const float r2 = x1 * sn + x2 * cs;

  if (z == 0) {
    bf16_t* dst = Qb + ((size_t)bh * S_SZ + s) * DK_SZ + 2 * p;
    dst[0] = (bf16_t)r1;
    dst[1] = (bf16_t)r2;
  } else {
    // transposed: Kbt[bh][d][s]
    Kbt[((size_t)bh * DK_SZ + 2 * p)     * S_SZ + s] = (bf16_t)r1;
    Kbt[((size_t)bh * DK_SZ + 2 * p + 1) * S_SZ + s] = (bf16_t)r2;
  }
}

// ---------------------------------------------------------------------------
// Kernel 4: causal flash attention, one (b,h) per blockIdx.y, 128 q-rows per
// block (8 waves x 16 rows). Online softmax; 32-key tiles staged in LDS.
// K arrives d-major; QK^T B-fragments are read d-strided from LDS.
// ---------------------------------------------------------------------------
__global__ __launch_bounds__(256) void flash_attn_kernel(
    const bf16_t* __restrict__ Qb, const bf16_t* __restrict__ Kbt,
    const bf16_t* __restrict__ Vb, bf16_t* __restrict__ Ob) {
  __shared__ __attribute__((aligned(16))) bf16_t Ktl[64][40];     // K^T[d][kv]
  __shared__ __attribute__((aligned(16))) bf16_t Vl[32][72];      // V[kv][d]
  __shared__ __attribute__((aligned(16))) bf16_t Pl[8][16][40];   // per-wave P

  const int tid   = threadIdx.x;
  const int lane  = tid & 31;
  const int wave  = tid >> 5;
  const int bh    = blockIdx.y;          // b*16 + h
  const int b     = bh >> 4;
  const int h     = bh & 15;
  const int q0    = blockIdx.x * 128;
  const int mlane = lane & 15;
  const int khalf = lane >> 4;
  const int qrow_base = q0 + wave * 16;

  // Q fragments: two 16x32 K-chunks of dk=64 (contiguous runs -> b128 loads)
  const bf16_t* qrow = Qb + ((size_t)bh * S_SZ + qrow_base + mlane) * DK_SZ;
  v16bf qa[2];
#pragma unroll
  for (int c = 0; c < 2; ++c)
#pragma unroll
    for (int j = 0; j < 8; ++j) {
      int k = c * 32 + (j < 4 ? 2 * j : 16 + 2 * (j - 4)) + khalf * 8;
      qa[c][2 * j]     = qrow[k];
      qa[c][2 * j + 1] = qrow[k + 1];
    }

  const v8f vzero = {0.f, 0.f, 0.f, 0.f, 0.f, 0.f, 0.f, 0.f};
  float mrow[8], lrow[8];
  v8f o[4];
#pragma unroll
  for (int j = 0; j < 8; ++j) { mrow[j] = -3.0e38f; lrow[j] = 0.f; }
#pragma unroll
  for (int nt = 0; nt < 4; ++nt) o[nt] = vzero;

  // K^T staging: thread copies 16B of one d-row (8 kv values)
  const int kd_row = tid >> 2;              // 0..63  (d)
  const int kd_seg = (tid & 3) * 8;         // kv segment
  const bf16_t* kptr = Kbt + ((size_t)bh * DK_SZ + kd_row) * S_SZ + kd_seg;
  // V staging: thread copies 16B of one kv-row (8 d values)
  const int kv_lane = tid >> 3;             // 0..31
  const int d_lane  = (tid & 7) * 8;
  const bf16_t* vptr = Vb + ((size_t)bh * S_SZ + kv_lane) * DK_SZ + d_lane;

  const int nblocks = q0 / 32 + 4;   // causal: keys up to q0+127
  for (int kb = 0; kb < nblocks; ++kb) {
    const int kv0 = kb * 32;
    __syncthreads();
    async_copy16(kptr + kv0, &Ktl[kd_row][kd_seg]);
    async_copy16(vptr + (size_t)kv0 * DK_SZ, &Vl[kv_lane][d_lane]);
    __builtin_prefetch(kptr + kv0 + 32, 0, 1);
    __builtin_prefetch(vptr + (size_t)(kv0 + 32) * DK_SZ, 0, 1);
    async_wait0();
    __syncthreads();

    // S = Q K^T : B-frag lane l -> kv = l&15 ; d in A-frag order
    v8f sfr[2] = {vzero, vzero};
#pragma unroll
    for (int c = 0; c < 2; ++c) {
#pragma unroll
      for (int nt = 0; nt < 2; ++nt) {
        v16bf bk = load_b_frag(&Ktl[c * 32][nt * 16 + mlane], 40, khalf);
        sfr[nt] = wmma_bf16(qa[c], bk, sfr[nt]);
      }
    }

    // scale, causal mask, online softmax
    float sv[2][8], bmax[8];
#pragma unroll
    for (int j = 0; j < 8; ++j) bmax[j] = -3.0e38f;
#pragma unroll
    for (int nt = 0; nt < 2; ++nt) {
      const int kcol = kv0 + nt * 16 + mlane;
#pragma unroll
      for (int j = 0; j < 8; ++j) {
        const int qr = qrow_base + khalf * 8 + j;
        float s = sfr[nt][j] * 0.125f;          // 1/sqrt(64)
        if (kcol > qr) s = -3.0e38f;
        sv[nt][j] = s;
        bmax[j] = fmaxf(bmax[j], s);
      }
    }
#pragma unroll
    for (int mask = 1; mask < 16; mask <<= 1)
#pragma unroll
      for (int j = 0; j < 8; ++j)
        bmax[j] = fmaxf(bmax[j], __shfl_xor(bmax[j], mask, 32));

    float corr[8], rsum[8];
#pragma unroll
    for (int j = 0; j < 8; ++j) {
      const float mn = fmaxf(mrow[j], bmax[j]);
      corr[j] = __expf(mrow[j] - mn);
      mrow[j] = mn;
      rsum[j] = 0.f;
    }
#pragma unroll
    for (int nt = 0; nt < 2; ++nt)
#pragma unroll
      for (int j = 0; j < 8; ++j) {
        const float pv = __expf(sv[nt][j] - mrow[j]);
        sv[nt][j] = pv;
        rsum[j] += pv;
      }
#pragma unroll
    for (int mask = 1; mask < 16; mask <<= 1)
#pragma unroll
      for (int j = 0; j < 8; ++j) rsum[j] += __shfl_xor(rsum[j], mask, 32);
#pragma unroll
    for (int j = 0; j < 8; ++j) lrow[j] = lrow[j] * corr[j] + rsum[j];
#pragma unroll
    for (int nt = 0; nt < 4; ++nt)
#pragma unroll
      for (int j = 0; j < 8; ++j) o[nt][j] *= corr[j];

    // P (C-layout) -> LDS -> A-fragment layout
#pragma unroll
    for (int nt = 0; nt < 2; ++nt)
#pragma unroll
      for (int j = 0; j < 8; ++j)
        Pl[wave][khalf * 8 + j][nt * 16 + mlane] = (bf16_t)sv[nt][j];
    __syncthreads();

    v16bf pa = load_a_frag(&Pl[wave][mlane][0], khalf);
#pragma unroll
    for (int nt = 0; nt < 4; ++nt) {
      v16bf bv = load_b_frag(&Vl[0][nt * 16 + mlane], 72, khalf);  // lane l -> d=l&15
      o[nt] = wmma_bf16(pa, bv, o[nt]);
    }
  }

  // epilogue: normalize and store to [b][s][d] bf16
  float rinv[8];
#pragma unroll
  for (int j = 0; j < 8; ++j) rinv[j] = 1.0f / lrow[j];
#pragma unroll
  for (int nt = 0; nt < 4; ++nt)
#pragma unroll
    for (int j = 0; j < 8; ++j) {
      const int srow = qrow_base + khalf * 8 + j;
      Ob[((size_t)b * S_SZ + srow) * D_SZ + h * DK_SZ + nt * 16 + mlane] =
          (bf16_t)(o[nt][j] * rinv[j]);
    }
}

// ---------------------------------------------------------------------------
// Launch
// ---------------------------------------------------------------------------
extern "C" void kernel_launch(void* const* d_in, const int* in_sizes, int n_in,
                              void* d_out, int out_size, void* d_ws, size_t ws_size,
                              hipStream_t stream) {
  const float* x   = (const float*)d_in[0];
  const int*   pos = (const int*)d_in[1];
  const float* Wq  = (const float*)d_in[2];
  const float* Wk  = (const float*)d_in[3];
  const float* Wv  = (const float*)d_in[4];
  const float* Wo  = (const float*)d_in[5];
  float* out = (float*)d_out;

  char* ws = (char*)d_ws;
  const size_t MB = 1024 * 1024;
  bf16_t* xb   = (bf16_t*)(ws + 0);        //  8 MB  [4096][1024]
  bf16_t* Wtb  = (bf16_t*)(ws + 8 * MB);   //  8 MB  [4][k=1024][n=1024]
  float*  QKVf = (float* )(ws + 16 * MB);  // 48 MB  [3][4096][1024]
  bf16_t* Qb   = (bf16_t*)(ws + 64 * MB);  //  8 MB  [b][h][s][dk]
  bf16_t* Kbt  = (bf16_t*)(ws + 72 * MB);  //  8 MB  [b][h][dk][s]
  bf16_t* Vb   = (bf16_t*)(ws + 80 * MB);  //  8 MB  [b][h][s][dk]
  bf16_t* Ob   = (bf16_t*)(ws + 88 * MB);  //  8 MB  [4096][1024]

  const int nX = M_TOT * D_SZ;      // 4194304
  const int nW = D_SZ * D_SZ;       // 1048576

  cvt_f32_bf16_kernel<<<nX / 4 / 256, 256, 0, stream>>>(x, xb, nX / 4);
  cvt_w_transpose_kernel<<<nW / 4 / 256, 256, 0, stream>>>(Wq, Wtb + 0 * (size_t)nW);
  cvt_w_transpose_kernel<<<nW / 4 / 256, 256, 0, stream>>>(Wk, Wtb + 1 * (size_t)nW);
  cvt_w_transpose_kernel<<<nW / 4 / 256, 256, 0, stream>>>(Wv, Wtb + 2 * (size_t)nW);
  cvt_w_transpose_kernel<<<nW / 4 / 256, 256, 0, stream>>>(Wo, Wtb + 3 * (size_t)nW);

  // Q/K/V projections: grid.z picks the weight + output slice
  gemm_bf16_kernel<<<dim3(D_SZ / 128, M_TOT / 128, 3), 256, 0, stream>>>(
      xb, Wtb, QKVf);

  // RoPE + head-major repack (2^21 pairs per tensor)
  rope_pack_kernel<<<dim3((M_TOT * D_SZ / 2) / 256, 3), 256, 0, stream>>>(
      QKVf, pos, Qb, Kbt, Vb);

  // causal flash attention
  flash_attn_kernel<<<dim3(S_SZ / 128, B_SZ * H_SZ), 256, 0, stream>>>(
      Qb, Kbt, Vb, Ob);

  // output projection -> fp32 d_out
  gemm_bf16_kernel<<<dim3(D_SZ / 128, M_TOT / 128, 1), 256, 0, stream>>>(
      Ob, Wtb + 3 * (size_t)nW, out);
}
